// MinamoTopoModel_3384434229898
// MI455X (gfx1250) — matrix operand order repaired and checked by reading, not compiled, not run on hardware
//
#include <hip/hip_runtime.h>

typedef __attribute__((ext_vector_type(16))) _Float16 v16h;
typedef __attribute__((ext_vector_type(8)))  _Float16 h8;
typedef __attribute__((ext_vector_type(8)))  float    v8f;
typedef __attribute__((ext_vector_type(4)))  unsigned int u32x4;
typedef __attribute__((ext_vector_type(4)))  int      i32x4;
typedef __attribute__((ext_vector_type(8)))  int      i32x8;

static constexpr int  NN   = 50000;
static constexpr int  EE   = 800000;
static constexpr int  GG   = 50;
static constexpr float EPSF = 1e-5f;

// ---------------- workspace layout (bytes) ----------------
static constexpr size_t OFF_F16A = 0;                                   // [N,512] f16 layer input
static constexpr size_t OFF_PROJ = OFF_F16A + (size_t)NN * 512 * 2;     // [N,512] f16 projected
static constexpr size_t OFF_AGG  = OFF_PROJ + (size_t)NN * 512 * 2;     // [N,512] f32 aggregation
static constexpr size_t OFF_WT   = OFF_AGG  + (size_t)NN * 512 * 4;     // [512,512] f16 W^T
static constexpr size_t OFF_AL   = OFF_WT   + (size_t)512 * 512 * 2;    // [N,8] f32
static constexpr size_t OFF_AR   = OFF_AL   + (size_t)NN * 8 * 4;
static constexpr size_t OFF_M    = OFF_AR   + (size_t)NN * 8 * 4;       // segment max
static constexpr size_t OFF_S    = OFF_M    + (size_t)NN * 8 * 4;       // segment sum
static constexpr size_t OFF_EB   = OFF_S    + (size_t)NN * 8 * 4;       // [(E+N),8] edge logits/exp
static constexpr size_t OFF_H3   = OFF_EB   + (size_t)(EE + NN) * 8 * 4;// [N,16] f32 final node feats
static constexpr size_t OFF_POOL = OFF_H3   + (size_t)NN * 16 * 4;      // [G,16]
static constexpr size_t OFF_CNT  = OFF_POOL + 4096;                     // [G]

__device__ __forceinline__ void atomicMaxF(float* a, float v) {
  if (v >= 0.f) atomicMax((int*)a, __float_as_int(v));
  else          atomicMin((unsigned int*)a, __float_as_uint(v));
}

// ---------------- utility kernels ----------------
__global__ void fill_f32(float* p, float v, long n) {
  long i = (long)blockIdx.x * blockDim.x + threadIdx.x;
  if (i < n) p[i] = v;
}

__global__ void embed_lookup(const int* __restrict__ x, const float* __restrict__ emb,
                             _Float16* __restrict__ h0, int n) {
  int i = blockIdx.x * blockDim.x + threadIdx.x;
  if (i >= n) return;
  int t = x[i];
  #pragma unroll
  for (int c = 0; c < 16; ++c) h0[(long)i * 32 + c] = (_Float16)emb[t * 16 + c];
  #pragma unroll
  for (int c = 16; c < 32; ++c) h0[(long)i * 32 + c] = (_Float16)0.f;
}

// W[k][o] (row-major [K,OUT]) -> Wt[o][k] f16, zero-padded to ldb
__global__ void prep_wt(const float* __restrict__ W, _Float16* __restrict__ Wt,
                        int K, int OUT, int ldb) {
  int i = blockIdx.x * blockDim.x + threadIdx.x;
  if (i >= OUT * ldb) return;
  int o = i / ldb, k = i % ldb;
  Wt[i] = (_Float16)((k < K) ? W[(long)k * OUT + o] : 0.f);
}

// ---------------- TDM: stage a 16xKP f16 tile into LDS (with bank-stagger padding) ----
// D# per CDNA5 ISA ch.8: 2D tensor, data_size=2B, pad 4 dwords after every row (KP/2 dwords).
template <int KP>
__device__ __forceinline__ void tdm_load_tileA(const _Float16* gsrc, _Float16* lds) {
  constexpr unsigned PADI = (KP == 512) ? 7u : 3u;   // pad interval: 256 / 16 dwords
  unsigned long long ga = (unsigned long long)(const void*)gsrc;
  unsigned ldsoff = (unsigned)(unsigned long long)(void*)lds;   // addr[31:0] = LDS offset
  u32x4 g0;
  g0[0] = 1u;                                        // count=1 (valid user descriptor)
  g0[1] = ldsoff;                                    // lds_addr
  g0[2] = (unsigned)ga;                              // global_addr[31:0]
  g0[3] = (unsigned)((ga >> 32) & 0x1FFFFFFull) | (2u << 30);   // global_addr[56:32] | type=2
  i32x8 g1;
  g1[0] = (int)((1u << 16) | (1u << 20) | (PADI << 22) | (3u << 25));
          // data_size=1(2B), pad_enable=1, pad_interval=PADI, pad_amount=3(4 dwords)
  g1[1] = (int)(((unsigned)KP & 0xFFFFu) << 16);     // tensor_dim0[15:0]
  g1[2] = (int)((((unsigned)KP >> 16) & 0xFFFFu) | (16u << 16)); // td0 hi | tensor_dim1=16
  g1[3] = (int)((unsigned)KP << 16);                 // tile_dim0 = KP
  g1[4] = 16;                                        // tile_dim1 = 16
  g1[5] = KP;                                        // tensor_dim0_stride[31:0]
  g1[6] = 0;
  g1[7] = 0;
  i32x4 gz4 = {0, 0, 0, 0};                          // groups 2/3 unused (2D)
#if __clang_major__ >= 23
  i32x8 gz8 = {};
  __builtin_amdgcn_tensor_load_to_lds(g0, g1, gz4, gz4, gz8, 0);
#else
  __builtin_amdgcn_tensor_load_to_lds(g0, g1, gz4, gz4, 0);
#endif
  __builtin_amdgcn_s_wait_tensorcnt(0);
}

// ---------------- WMMA GEMM: D[N,OUT] = A[N,KP] * Wt[OUT,KP]^T ----------------
// A tile staged once per workgroup via TDM; B streamed from global (Wt lives in L2).
template <int KP, int OUT>
__global__ __launch_bounds__(256) void gemm_wmma_f16(const _Float16* __restrict__ A,
                                                     const _Float16* __restrict__ Bt,
                                                     _Float16* __restrict__ D) {
  constexpr int ROWB = KP + 8;                       // padded LDS row (halves): +16B stagger
  __shared__ _Float16 sA[16 * ROWB];
  if (threadIdx.x < 32)                              // wave 0 issues the TDM op + waits
    tdm_load_tileA<KP>(A + (size_t)blockIdx.x * 16 * KP, sA);
  __syncthreads();

  const int lane = threadIdx.x & 31;
  const int wv   = threadIdx.x >> 5;
  const int colTile = blockIdx.y * 8 + wv;
  if (colTile * 16 < OUT) {
    const int mr = lane & 15;                        // row-in-tile (A) / col-in-tile (B,D)
    const int hl = lane >> 4;                        // lane-half selects K sub-range
    const _Float16* ap = sA + mr * ROWB;
    const _Float16* bp = Bt + ((long)colTile * 16 + mr) * KP;
    v8f acc = {};
    #pragma unroll
    for (int k0 = 0; k0 < KP; k0 += 32) {
      if (k0 + 64 < KP) __builtin_prefetch(bp + k0 + 64, 0, 0);
      // A 16x32 frag: elems 0..7 = K[k0+8*hl..+7], elems 8..15 = K[k0+16+8*hl..+7]
      union { v16h v; h8 h[2]; } ua;
      ua.h[0] = *(const h8*)(ap + k0 + 8 * hl);
      ua.h[1] = *(const h8*)(ap + k0 + 16 + 8 * hl);
      // B 32x16 frag: elems 0..7 = K[k0+8*hl..+7], elems 8..15 = K[k0+16+8*hl..+7] of column mr (= row mr of Wt tile)
      union { v16h v; h8 h[2]; } ub;
      ub.h[0] = *(const h8*)(bp + k0 + 8 * hl);
      ub.h[1] = *(const h8*)(bp + k0 + 16 + 8 * hl);
      acc = __builtin_amdgcn_wmma_f32_16x16x32_f16(false, ua.v, false, ub.v,
                                                   (short)0, acc, false, false);
    }
    // D layout: VGPR r -> M = r + 8*hl, N = mr ; OUT constexpr -> immediate store offsets
    _Float16* dp = D + ((long)blockIdx.x * 16 + 8 * hl) * OUT + colTile * 16 + mr;
    #pragma unroll
    for (int r = 0; r < 8; ++r) dp[r * OUT] = (_Float16)acc[r];
  }
}

// ---------------- attention scalars: al/ar per (node, head) ----------------
__global__ void node_alar(const _Float16* __restrict__ proj,
                          const float* __restrict__ a_src, const float* __restrict__ a_dst,
                          float* __restrict__ al, float* __restrict__ ar,
                          int n, int H, int C) {
  int i = blockIdx.x * blockDim.x + threadIdx.x;
  if (i >= n * H) return;
  int node = i / H, h = i % H;
  const _Float16* p = proj + (long)node * H * C + h * C;
  const float* as = a_src + h * C;
  const float* ad = a_dst + h * C;
  float sl = 0.f, sr = 0.f;
  for (int c = 0; c < C; ++c) { float v = (float)p[c]; sl += v * as[c]; sr += v * ad[c]; }
  al[i] = sl; ar[i] = sr;
}

__device__ __forceinline__ void edge_decode(long j, int E_, const int* es, const int* ed,
                                            int& s, int& d) {
  if (j < E_) { s = es[j]; d = ed[j]; } else { s = d = (int)(j - E_); }
}

__global__ void edge_logits(const int* __restrict__ es, const int* __restrict__ ed,
                            int E_, int n, const float* __restrict__ al,
                            const float* __restrict__ ar, float* __restrict__ ebuf,
                            float* __restrict__ mbuf, int H) {
  long i = (long)blockIdx.x * blockDim.x + threadIdx.x;
  if (i >= (long)(E_ + n) * H) return;
  int h = (int)(i % H); long j = i / H;
  int s, d; edge_decode(j, E_, es, ed, s, d);
  float e = al[(long)s * H + h] + ar[(long)d * H + h];
  e = e > 0.f ? e : 0.2f * e;                       // leaky_relu(0.2)
  ebuf[i] = e;
  atomicMaxF(mbuf + (long)d * H + h, e);
}

__global__ void edge_expsum(const int* __restrict__ es, const int* __restrict__ ed,
                            int E_, int n, float* __restrict__ ebuf,
                            const float* __restrict__ mbuf, float* __restrict__ ssum, int H) {
  long i = (long)blockIdx.x * blockDim.x + threadIdx.x;
  if (i >= (long)(E_ + n) * H) return;
  int h = (int)(i % H); long j = i / H;
  int s, d; edge_decode(j, E_, es, ed, s, d);
  (void)s;
  float ex = __expf(ebuf[i] - mbuf[(long)d * H + h]);
  ebuf[i] = ex;
  atomicAdd(ssum + (long)d * H + h, ex);
}

// wave per edge: agg[dst][ch] += proj[src][ch] * ex[edge][head(ch)]
template <int OUT>
__global__ void edge_aggregate(const int* __restrict__ es, const int* __restrict__ ed,
                               int E_, int n, const _Float16* __restrict__ proj,
                               const float* __restrict__ exbuf, float* __restrict__ agg,
                               int H, int lc) {
  int lane = threadIdx.x & 31;
  long j = (long)blockIdx.x * (blockDim.x >> 5) + (threadIdx.x >> 5);
  if (j >= (long)(E_ + n)) return;
  int s, d; edge_decode(j, E_, es, ed, s, d);
  if constexpr (OUT == 512) {
    // lane owns 16 contiguous channels; head index constant within the chunk
    const h8* ps = (const h8*)(proj + (long)s * OUT + lane * 16);
    h8 a0 = ps[0], a1 = ps[1];                       // two global_load_b128
    float* pd = agg + (long)d * OUT + lane * 16;
    float ex = exbuf[j * H + ((lane * 16) >> lc)];
    #pragma unroll
    for (int i = 0; i < 8; ++i) atomicAdd(pd + i, (float)a0[i] * ex);
    #pragma unroll
    for (int i = 0; i < 8; ++i) atomicAdd(pd + 8 + i, (float)a1[i] * ex);
  } else {
    for (int ch = lane; ch < OUT; ch += 32)
      atomicAdd(agg + (long)d * OUT + ch,
                (float)proj[(long)s * OUT + ch] * exbuf[j * H + (ch >> lc)]);
  }
}

// ---------------- finalize: /softmax-sum, +bias, LayerNorm, ELU -> f16 ----------------
template <int OUT, int H, int LC>
__global__ void finalize_concat(const float* __restrict__ agg, const float* __restrict__ ssum,
                                const float* __restrict__ bias, const float* __restrict__ g,
                                const float* __restrict__ be, _Float16* __restrict__ hout,
                                int n) {
  int lane = threadIdx.x & 31;
  long node = (long)blockIdx.x * (blockDim.x >> 5) + (threadIdx.x >> 5);
  if (node >= n) return;
  constexpr int PER = OUT / 32;
  float v[PER];
  float sum = 0.f;
  #pragma unroll
  for (int i = 0; i < PER; ++i) {
    int ch = lane + 32 * i;
    float x = agg[node * OUT + ch] / ssum[node * H + (ch >> LC)] + bias[ch];
    v[i] = x; sum += x;
  }
  #pragma unroll
  for (int o = 16; o; o >>= 1) sum += __shfl_xor(sum, o, 32);
  float mean = sum / OUT;
  float var = 0.f;
  #pragma unroll
  for (int i = 0; i < PER; ++i) { float d = v[i] - mean; var += d * d; }
  #pragma unroll
  for (int o = 16; o; o >>= 1) var += __shfl_xor(var, o, 32);
  float inv = rsqrtf(var / OUT + EPSF);
  #pragma unroll
  for (int i = 0; i < PER; ++i) {
    int ch = lane + 32 * i;
    float y = (v[i] - mean) * inv * g[ch] + be[ch];
    y = y > 0.f ? y : (__expf(y) - 1.f);            // ELU
    hout[node * OUT + ch] = (_Float16)y;
  }
}

// layer 3: OUT=16, H=1, write f32
__global__ void finalize16(const float* __restrict__ agg, const float* __restrict__ ssum,
                           const float* __restrict__ bias, const float* __restrict__ g,
                           const float* __restrict__ be, float* __restrict__ h3, int n) {
  int lane = threadIdx.x & 31;
  long node = (long)blockIdx.x * (blockDim.x >> 5) + (threadIdx.x >> 5);
  if (node >= n) return;
  bool act = lane < 16;
  float x = act ? (agg[node * 16 + lane] / ssum[node] + bias[lane]) : 0.f;
  float sum = x;
  #pragma unroll
  for (int o = 16; o; o >>= 1) sum += __shfl_xor(sum, o, 32);
  float mean = sum / 16.f;
  float dd = act ? (x - mean) : 0.f;
  float var = dd * dd;
  #pragma unroll
  for (int o = 16; o; o >>= 1) var += __shfl_xor(var, o, 32);
  float inv = rsqrtf(var / 16.f + EPSF);
  if (act) {
    float y = (x - mean) * inv * g[lane] + be[lane];
    y = y > 0.f ? y : (__expf(y) - 1.f);
    h3[node * 16 + lane] = y;
  }
}

// ---------------- pooling + FC head ----------------
__global__ void pool_sum(const float* __restrict__ h3, const int* __restrict__ batch,
                         float* __restrict__ pool, float* __restrict__ cnt, int n) {
  int i = blockIdx.x * blockDim.x + threadIdx.x;
  if (i >= n) return;
  int b = batch[i];
  #pragma unroll
  for (int c = 0; c < 16; ++c) atomicAdd(pool + b * 16 + c, h3[(long)i * 16 + c]);
  atomicAdd(cnt + b, 1.f);
}

__global__ void fc_head(const float* __restrict__ pool, const float* __restrict__ cnt,
                        const float* __restrict__ W1, const float* __restrict__ b1,
                        const float* __restrict__ W2, const float* __restrict__ b2,
                        float* __restrict__ out) {
  int g = threadIdx.x;
  if (g >= GG) return;
  float p[16], t[16];
  float c = fmaxf(cnt[g], 1.f);
  #pragma unroll
  for (int i = 0; i < 16; ++i) p[i] = pool[g * 16 + i] / c;
  #pragma unroll
  for (int j = 0; j < 16; ++j) {
    float a = b1[j];
    #pragma unroll
    for (int i = 0; i < 16; ++i) a += p[i] * W1[i * 16 + j];
    t[j] = fmaxf(a, 0.f);
  }
  #pragma unroll
  for (int k = 0; k < 8; ++k) {
    float a = b2[k];
    #pragma unroll
    for (int j = 0; j < 16; ++j) a += t[j] * W2[j * 8 + k];
    out[g * 8 + k] = a;
  }
}

// ---------------- host orchestration ----------------
static inline int cdiv(long a, long b) { return (int)((a + b - 1) / b); }

extern "C" void kernel_launch(void* const* d_in, const int* in_sizes, int n_in,
                              void* d_out, int out_size, void* d_ws, size_t ws_size,
                              hipStream_t stream) {
  const int*   x     = (const int*)d_in[0];
  const int*   eidx  = (const int*)d_in[1];
  const int*   batch = (const int*)d_in[2];
  const float* emb   = (const float*)d_in[3];
  const float* W1  = (const float*)d_in[4];
  const float* as1 = (const float*)d_in[5];
  const float* ad1 = (const float*)d_in[6];
  const float* b1  = (const float*)d_in[7];
  const float* g1  = (const float*)d_in[8];
  const float* be1 = (const float*)d_in[9];
  const float* W2  = (const float*)d_in[10];
  const float* as2 = (const float*)d_in[11];
  const float* ad2 = (const float*)d_in[12];
  const float* b2  = (const float*)d_in[13];
  const float* g2  = (const float*)d_in[14];
  const float* be2 = (const float*)d_in[15];
  const float* W3  = (const float*)d_in[16];
  const float* as3 = (const float*)d_in[17];
  const float* ad3 = (const float*)d_in[18];
  const float* b3  = (const float*)d_in[19];
  const float* g3  = (const float*)d_in[20];
  const float* be3 = (const float*)d_in[21];
  const float* fcW1 = (const float*)d_in[22];
  const float* fcb1 = (const float*)d_in[23];
  const float* fcW2 = (const float*)d_in[24];
  const float* fcb2 = (const float*)d_in[25];
  float* out = (float*)d_out;

  char* ws = (char*)d_ws;
  _Float16* f16A = (_Float16*)(ws + OFF_F16A);
  _Float16* proj = (_Float16*)(ws + OFF_PROJ);
  float*    agg  = (float*)(ws + OFF_AGG);
  _Float16* Wt   = (_Float16*)(ws + OFF_WT);
  float*    al   = (float*)(ws + OFF_AL);
  float*    ar   = (float*)(ws + OFF_AR);
  float*    mbuf = (float*)(ws + OFF_M);
  float*    sbuf = (float*)(ws + OFF_S);
  float*    ebuf = (float*)(ws + OFF_EB);
  float*    h3   = (float*)(ws + OFF_H3);
  float*    pool = (float*)(ws + OFF_POOL);
  float*    cnt  = (float*)(ws + OFF_CNT);

  const int* esrc = eidx;
  const int* edst = eidx + EE;
  const int ETOT = EE + NN;

  // ---- embedding -> f16A [N,32] (K=16 zero-padded to 32) ----
  embed_lookup<<<cdiv(NN, 256), 256, 0, stream>>>(x, emb, f16A, NN);

  // common per-layer attention scalar passes (logits / softmax-max / exp-sum)
  auto attn_scalars = [&](const float* asr, const float* adr, int OUT, int H, int C) {
    node_alar<<<cdiv((long)NN * H, 256), 256, 0, stream>>>(proj, asr, adr, al, ar, NN, H, C);
    fill_f32<<<cdiv((long)NN * H, 256), 256, 0, stream>>>(mbuf, -3.4e38f, (long)NN * H);
    fill_f32<<<cdiv((long)NN * H, 256), 256, 0, stream>>>(sbuf, 0.f, (long)NN * H);
    fill_f32<<<cdiv((long)NN * OUT, 256), 256, 0, stream>>>(agg, 0.f, (long)NN * OUT);
    long tot = (long)ETOT * H;
    edge_logits<<<cdiv(tot, 256), 256, 0, stream>>>(esrc, edst, EE, NN, al, ar, ebuf, mbuf, H);
    edge_expsum<<<cdiv(tot, 256), 256, 0, stream>>>(esrc, edst, EE, NN, ebuf, mbuf, sbuf, H);
  };

  // ---- layer 1: 16(pad32) -> 8x64 = 512 ----
  prep_wt<<<cdiv(512 * 32, 256), 256, 0, stream>>>(W1, Wt, 16, 512, 32);
  gemm_wmma_f16<32, 512><<<dim3(NN / 16, 4), 256, 0, stream>>>(f16A, Wt, proj);
  attn_scalars(as1, ad1, 512, 8, 64);
  edge_aggregate<512><<<cdiv(ETOT, 8), 256, 0, stream>>>(esrc, edst, EE, NN, proj, ebuf, agg, 8, 6);
  finalize_concat<512, 8, 6><<<cdiv(NN, 8), 256, 0, stream>>>(agg, sbuf, b1, g1, be1, f16A, NN);

  // ---- layer 2: 512 -> 4x128 = 512 ----
  prep_wt<<<cdiv(512 * 512, 256), 256, 0, stream>>>(W2, Wt, 512, 512, 512);
  gemm_wmma_f16<512, 512><<<dim3(NN / 16, 4), 256, 0, stream>>>(f16A, Wt, proj);
  attn_scalars(as2, ad2, 512, 4, 128);
  edge_aggregate<512><<<cdiv(ETOT, 8), 256, 0, stream>>>(esrc, edst, EE, NN, proj, ebuf, agg, 4, 7);
  finalize_concat<512, 4, 7><<<cdiv(NN, 8), 256, 0, stream>>>(agg, sbuf, b2, g2, be2, f16A, NN);

  // ---- layer 3: 512 -> 1x16 = 16 (mean over 1 head) ----
  prep_wt<<<cdiv(16 * 512, 256), 256, 0, stream>>>(W3, Wt, 512, 16, 512);
  gemm_wmma_f16<512, 16><<<dim3(NN / 16, 1), 32, 0, stream>>>(f16A, Wt, proj);
  attn_scalars(as3, ad3, 16, 1, 16);
  edge_aggregate<16><<<cdiv(ETOT, 8), 256, 0, stream>>>(esrc, edst, EE, NN, proj, ebuf, agg, 1, 4);
  finalize16<<<cdiv(NN, 8), 256, 0, stream>>>(agg, sbuf, b3, g3, be3, h3, NN);

  // ---- mean pool per graph + FC head ----
  fill_f32<<<cdiv(GG * 16, 256), 256, 0, stream>>>(pool, 0.f, GG * 16);
  fill_f32<<<1, 256, 0, stream>>>(cnt, 0.f, GG);
  pool_sum<<<cdiv(NN, 256), 256, 0, stream>>>(h3, batch, pool, cnt, NN);
  fc_head<<<1, 64, 0, stream>>>(pool, cnt, fcW1, fcb1, fcW2, fcb2, out);

  (void)in_sizes; (void)n_in; (void)out_size; (void)ws_size;
}
